// SimpleSparseAttention_74105365725867
// MI455X (gfx1250) — matrix operand order repaired and checked by reading, not compiled, not run on hardware
//
#include <hip/hip_runtime.h>
#include <hip/hip_bf16.h>

typedef __attribute__((ext_vector_type(16))) _Float16 v16h;
typedef __attribute__((ext_vector_type(8)))  _Float16 v8h;
typedef __attribute__((ext_vector_type(8)))  float    v8f;

#define NT   2048   // tokens
#define DM   512    // model dim
#define NH   8      // heads
#define HDM  64     // head dim
#define CHK  64     // chunk size
#define NG   32     // num chunks
#define DG   16     // 2*H gate logits
#define SCL  0.125f // 1/sqrt(64)
#define NEG_INF (-__builtin_inff())

union V16U { v16h v; v8h h[2]; };

// ---- WMMA fragment helpers (layouts per CDNA5 ISA §7.12.2) ----

// A: 16x32 f16, row-major source [M, K], leading dim lda.
__device__ __forceinline__ v16h load_a_frag(const _Float16* A, int lda, int m0, int k0) {
  int lane = threadIdx.x & 31;
  int m = m0 + (lane & 15);
  int sel = lane >> 4;
  const _Float16* p = A + (size_t)m * lda + k0 + sel * 8;
  V16U r;
  r.h[0] = *(const v8h*)(p);
  r.h[1] = *(const v8h*)(p + 16);
  return r.v;
}

// B: 32x16 f16 (K x N), sourced from row-major Bt[N, K] (B = Bt^T).
__device__ __forceinline__ v16h load_b_frag(const _Float16* Bt, int ldb, int n0, int k0) {
  int lane = threadIdx.x & 31;
  int n = n0 + (lane & 15);
  int sel = lane >> 4;
  const _Float16* p = Bt + (size_t)n * ldb + k0 + sel * 8;
  V16U r;
  r.h[0] = *(const v8h*)(p);
  r.h[1] = *(const v8h*)(p + 16);
  return r.v;
}

__device__ __forceinline__ v8f wmma16(v16h a, v16h b, v8f c) {
  return __builtin_amdgcn_wmma_f32_16x16x32_f16(
      false, a, false, b, (short)0, c, false, false);
}

// ---- CDNA5 async global->LDS copy (ASYNCcnt path, §15.18 opcode 98) ----
// GVS form: LDS byte-addr in VGPR, 32-bit byte offset in VGPR, base in SGPRs.
__device__ __forceinline__ void async_g2l_b128(unsigned lds_addr, unsigned goff,
                                               const void* base) {
  asm volatile("global_load_async_to_lds_b128 %0, %1, %2"
               :
               : "v"(lds_addr), "v"(goff), "s"(base)
               : "memory");
}
__device__ __forceinline__ void wait_asynccnt0() {
  asm volatile("s_wait_asynccnt 0x0" ::: "memory");
}
// low 32 bits of a generic pointer to LDS == LDS byte offset (flat aperture)
__device__ __forceinline__ unsigned lds_offset_of(const void* p) {
  return (unsigned)(size_t)p;
}

// ---- fp32 -> fp16 staging ----
__global__ void cvt_f16_kernel(const float* __restrict__ in,
                               _Float16* __restrict__ out, int n) {
  int t = blockIdx.x * blockDim.x + threadIdx.x;
  if (t < n) out[t] = (_Float16)in[t];
}

// ---- 16x64-strip GEMM: C[M,N] = A[M,K] * Bt[N,K]^T, 4 WMMA tiles / A-frag ----
template <typename OutT>
__global__ void gemm4_kernel(const _Float16* __restrict__ A,
                             const _Float16* __restrict__ Bt,
                             OutT* __restrict__ C, int M, int N, int K) {
  int wave = (blockIdx.x * blockDim.x + threadIdx.x) >> 5;  // wave-uniform
  int strips = N >> 6;
  int m0 = (wave / strips) << 4;
  int n0 = (wave % strips) << 6;
  if (m0 >= M) return;  // whole-wave uniform
  v8f a0 = {}, a1 = {}, a2 = {}, a3 = {};
  for (int k0 = 0; k0 < K; k0 += 32) {
    v16h a  = load_a_frag(A, K, m0, k0);
    v16h b0 = load_b_frag(Bt, K, n0,      k0);
    v16h b1 = load_b_frag(Bt, K, n0 + 16, k0);
    v16h b2 = load_b_frag(Bt, K, n0 + 32, k0);
    v16h b3 = load_b_frag(Bt, K, n0 + 48, k0);
    a0 = wmma16(a, b0, a0);
    a1 = wmma16(a, b1, a1);
    a2 = wmma16(a, b2, a2);
    a3 = wmma16(a, b3, a3);
  }
  int lane = threadIdx.x & 31;
  int nl = lane & 15;
  int mb = m0 + ((lane >> 4) << 3);
  v8f accs[4] = {a0, a1, a2, a3};
#pragma unroll
  for (int t = 0; t < 4; ++t) {
    int n = n0 + t * 16 + nl;
#pragma unroll
    for (int r = 0; r < 8; ++r) C[(size_t)(mb + r) * N + n] = (OutT)accs[t][r];
  }
}

// ---- single-tile GEMM (for narrow N, e.g. gate logits N=16) ----
__global__ void gemm1_kernel(const _Float16* __restrict__ A,
                             const _Float16* __restrict__ Bt,
                             float* __restrict__ C, int M, int N, int K) {
  int wave = (blockIdx.x * blockDim.x + threadIdx.x) >> 5;
  int tilesN = N >> 4;
  int m0 = (wave / tilesN) << 4;
  int n0 = (wave % tilesN) << 4;
  if (m0 >= M) return;
  v8f acc = {};
  for (int k0 = 0; k0 < K; k0 += 32) {
    v16h a = load_a_frag(A, K, m0, k0);
    v16h b = load_b_frag(Bt, K, n0, k0);
    acc = wmma16(a, b, acc);
  }
  int lane = threadIdx.x & 31;
  int n  = n0 + (lane & 15);
  int mb = m0 + ((lane >> 4) << 3);
#pragma unroll
  for (int r = 0; r < 8; ++r) C[(size_t)(mb + r) * N + n] = acc[r];
}

// ---- compressed keys: mean over each chunk (f16 in, f32 out) ----
__global__ void kcompress_kernel(const _Float16* __restrict__ k,
                                 float* __restrict__ kc) {
  int g = blockIdx.x;   // 0..31
  int d = threadIdx.x;  // 0..511
  const _Float16* p = k + (size_t)g * CHK * DM + d;
  float s = 0.f;
#pragma unroll 8
  for (int i = 0; i < CHK; ++i) s += (float)p[(size_t)i * DM];
  kc[g * DM + d] = s * (1.0f / (float)CHK);
}

// ---- per-(h,n) block scores + masked top-2 (wave per query, lane = block) ----
__global__ void select_top2_kernel(const _Float16* __restrict__ q,
                                   const float* __restrict__ kc,
                                   int* __restrict__ idx) {
  int gid  = (blockIdx.x * blockDim.x + threadIdx.x) >> 5;
  int lane = threadIdx.x & 31;  // lane == block g
  int h = gid >> 11;
  int n = gid & (NT - 1);
  const _Float16* qp = q + (size_t)n * DM + h * HDM;
  const float*    kp = kc + (size_t)lane * DM + h * HDM;
  float s = 0.f;
#pragma unroll 8
  for (int d = 0; d < HDM; ++d) s = fmaf((float)qp[d], kp[d], s);
  if (lane == (n >> 6)) s = NEG_INF;  // mask own block
  float v1 = s; int i1 = lane;
  for (int off = 16; off; off >>= 1) {
    float ov = __shfl_xor(v1, off, 32);
    int   oi = __shfl_xor(i1, off, 32);
    if (ov > v1 || (ov == v1 && oi < i1)) { v1 = ov; i1 = oi; }
  }
  float s2 = (lane == i1) ? NEG_INF : s;
  float v2 = s2; int i2 = lane;
  for (int off = 16; off; off >>= 1) {
    float ov = __shfl_xor(v2, off, 32);
    int   oi = __shfl_xor(i2, off, 32);
    if (ov > v2 || (ov == v2 && oi < i2)) { v2 = ov; i2 = oi; }
  }
  if (lane == 0) {
    idx[((size_t)h * NT + n) * 2 + 0] = i1;
    idx[((size_t)h * NT + n) * 2 + 1] = i2;
  }
}

// ---- inter attention: wave per (h,n), 2 gathered blocks = 128 keys ----
__global__ void inter_kernel(const _Float16* __restrict__ q,
                             const _Float16* __restrict__ k,
                             const _Float16* __restrict__ v,
                             const int* __restrict__ idx,
                             float* __restrict__ oi) {
  __shared__ float sc[8][128];
  int wid = threadIdx.x >> 5, lane = threadIdx.x & 31;
  int gid = (blockIdx.x * blockDim.x + threadIdx.x) >> 5;
  int h = gid >> 11, n = gid & (NT - 1);
  int b0 = idx[((size_t)h * NT + n) * 2 + 0];
  int b1 = idx[((size_t)h * NT + n) * 2 + 1];
  const _Float16* qp = q + (size_t)n * DM + h * HDM;
  for (int t = 0; t < 4; ++t) {
    int m = lane + t * 32;
    int blk = (m < 64) ? b0 : b1;
    const _Float16* kp = k + (size_t)(blk * CHK + (m & 63)) * DM + h * HDM;
    float s = 0.f;
#pragma unroll 8
    for (int d = 0; d < HDM; ++d) s = fmaf((float)qp[d], (float)kp[d], s);
    sc[wid][m] = s * SCL;
  }
  __syncthreads();
  float mx = NEG_INF;
  for (int t = 0; t < 4; ++t) mx = fmaxf(mx, sc[wid][lane + t * 32]);
  for (int off = 16; off; off >>= 1) mx = fmaxf(mx, __shfl_xor(mx, off, 32));
  float se = 0.f;
  for (int t = 0; t < 4; ++t) se += __expf(sc[wid][lane + t * 32] - mx);
  for (int off = 16; off; off >>= 1) se += __shfl_xor(se, off, 32);
  float inv = 1.f / se;
  float o0 = 0.f, o1 = 0.f;
  for (int j = 0; j < 128; ++j) {
    float w = __expf(sc[wid][j] - mx) * inv;
    int blk = (j < 64) ? b0 : b1;
    const _Float16* vp = v + (size_t)(blk * CHK + (j & 63)) * DM + h * HDM;
    o0 = fmaf(w, (float)vp[lane], o0);
    o1 = fmaf(w, (float)vp[lane + 32], o1);
  }
  oi[(size_t)n * DM + h * HDM + lane]      = o0;
  oi[(size_t)n * DM + h * HDM + lane + 32] = o1;
}

// ---- intra-chunk causal attention: block per (g,h) ----
// Q/K tiles staged via GLOBAL_LOAD_ASYNC_TO_LDS_B128; WMMA for both GEMMs.
__global__ void intra_kernel(const _Float16* __restrict__ q,
                             const _Float16* __restrict__ k,
                             const _Float16* __restrict__ v,
                             float* __restrict__ oa) {
  __shared__ __align__(16) _Float16 Qh[CHK * HDM];  // [i][d]
  __shared__ __align__(16) _Float16 Kh[CHK * HDM];  // [j][d] (Bt for S)
  __shared__ __align__(16) _Float16 Vt[HDM * CHK];  // [d][j] (Bt for O)
  __shared__ float Sf[CHK * CHK];
  __shared__ __align__(16) _Float16 Ah[CHK * CHK];
  int g = blockIdx.x >> 3, h = blockIdx.x & 7;
  size_t tile_byte = ((size_t)g * CHK * DM + h * HDM) * sizeof(_Float16);

  // async-stage Q and K: 64 rows x 128B, 16B per lane-op (512 chunks each)
  unsigned ldsQ = lds_offset_of(Qh);
  unsigned ldsK = lds_offset_of(Kh);
  for (int c = threadIdx.x; c < 512; c += 256) {
    int row = c >> 3, seg = c & 7;
    unsigned goff = (unsigned)(tile_byte + (size_t)(row * DM + seg * 8) * 2);
    unsigned loff = (unsigned)(row * HDM + seg * 8) * 2;
    async_g2l_b128(ldsQ + loff, goff, q);
    async_g2l_b128(ldsK + loff, goff, k);
  }
  // V staged transposed (VALU path; async can't transpose)
  const _Float16* vb = v + (size_t)g * CHK * DM + h * HDM;
  for (int e = threadIdx.x; e < CHK * HDM; e += 256) {
    int i = e >> 6, d = e & 63;
    Vt[d * CHK + i] = vb[(size_t)i * DM + d];
  }
  wait_asynccnt0();
  __syncthreads();

  int wid = threadIdx.x >> 5, lane = threadIdx.x & 31;
  // S = Q * K^T (16 tiles, 2 per wave)
  for (int t = wid; t < 16; t += 8) {
    int ti = t >> 2, tj = t & 3;
    v8f acc = {};
    for (int k0 = 0; k0 < HDM; k0 += 32) {
      v16h a = load_a_frag(Qh, HDM, ti * 16, k0);
      v16h b = load_b_frag(Kh, HDM, tj * 16, k0);
      acc = wmma16(a, b, acc);
    }
    int nn = tj * 16 + (lane & 15);
    int mb = ti * 16 + ((lane >> 4) << 3);
#pragma unroll
    for (int r = 0; r < 8; ++r) Sf[(mb + r) * CHK + nn] = acc[r] * SCL;
  }
  __syncthreads();
  // causal row softmax -> Ah (f16)
  if (threadIdx.x < CHK) {
    int r = threadIdx.x;
    float mx = NEG_INF;
    for (int j = 0; j <= r; ++j) mx = fmaxf(mx, Sf[r * CHK + j]);
    float se = 0.f;
    for (int j = 0; j <= r; ++j) se += __expf(Sf[r * CHK + j] - mx);
    float inv = 1.f / se;
    for (int j = 0; j < CHK; ++j)
      Ah[r * CHK + j] =
          (_Float16)((j <= r) ? __expf(Sf[r * CHK + j] - mx) * inv : 0.f);
  }
  __syncthreads();
  // O = A * V
  float* ob = oa + (size_t)g * CHK * DM + h * HDM;
  for (int t = wid; t < 16; t += 8) {
    int ti = t >> 2, tj = t & 3;
    v8f acc = {};
    for (int k0 = 0; k0 < CHK; k0 += 32) {
      v16h a = load_a_frag(Ah, CHK, ti * 16, k0);
      v16h b = load_b_frag(Vt, CHK, tj * 16, k0);
      acc = wmma16(a, b, acc);
    }
    int nn = tj * 16 + (lane & 15);
    int mb = ti * 16 + ((lane >> 4) << 3);
#pragma unroll
    for (int r = 0; r < 8; ++r) ob[(size_t)(mb + r) * DM + nn] = acc[r];
  }
}

// ---- gate (softmax over 2 == sigmoid) + combine + f16 convert ----
__global__ void combine_kernel(const float* __restrict__ oi,
                               const float* __restrict__ oa,
                               const float* __restrict__ gl,
                               _Float16* __restrict__ oh) {
  int t = blockIdx.x * blockDim.x + threadIdx.x;  // over N*D
  int n = t >> 9, d = t & (DM - 1), h = d >> 6;
  float l0 = gl[n * DG + h * 2 + 0];
  float l1 = gl[n * DG + h * 2 + 1];
  float g0 = 1.f / (1.f + __expf(l1 - l0));
  float val = g0 * oi[t] + (1.f - g0) * oa[t];
  oh[t] = (_Float16)val;
}

extern "C" void kernel_launch(void* const* d_in, const int* in_sizes, int n_in,
                              void* d_out, int out_size, void* d_ws, size_t ws_size,
                              hipStream_t stream) {
  const float* x  = (const float*)d_in[0];
  const float* Wq = (const float*)d_in[1];
  const float* Wk = (const float*)d_in[2];
  const float* Wv = (const float*)d_in[3];
  const float* Wg = (const float*)d_in[4];
  const float* Wo = (const float*)d_in[5];
  float* out = (float*)d_out;

  char* ws = (char*)d_ws;
  size_t off = 0;
  auto alloc = [&](size_t bytes) -> void* {
    void* p = ws + off;
    off = (off + bytes + 255) & ~(size_t)255;
    return p;
  };
  _Float16* xh  = (_Float16*)alloc((size_t)NT * DM * 2);
  _Float16* Wqh = (_Float16*)alloc((size_t)DM * DM * 2);
  _Float16* Wkh = (_Float16*)alloc((size_t)DM * DM * 2);
  _Float16* Wvh = (_Float16*)alloc((size_t)DM * DM * 2);
  _Float16* Woh = (_Float16*)alloc((size_t)DM * DM * 2);
  _Float16* Wgh = (_Float16*)alloc((size_t)DG * DM * 2);
  _Float16* qh  = (_Float16*)alloc((size_t)NT * DM * 2);
  _Float16* kh  = (_Float16*)alloc((size_t)NT * DM * 2);
  _Float16* vh  = (_Float16*)alloc((size_t)NT * DM * 2);
  float* gl = (float*)alloc((size_t)NT * DG * 4);
  float* kc = (float*)alloc((size_t)NG * DM * 4);
  int*   ix = (int*)alloc((size_t)NH * NT * 2 * 4);
  float* oi = (float*)alloc((size_t)NT * DM * 4);
  float* oa = (float*)alloc((size_t)NT * DM * 4);
  _Float16* oh = (_Float16*)alloc((size_t)NT * DM * 2);

  // stage fp32 -> fp16
  cvt_f16_kernel<<<(NT * DM) / 256, 256, 0, stream>>>(x,  xh,  NT * DM);
  cvt_f16_kernel<<<(DM * DM) / 256, 256, 0, stream>>>(Wq, Wqh, DM * DM);
  cvt_f16_kernel<<<(DM * DM) / 256, 256, 0, stream>>>(Wk, Wkh, DM * DM);
  cvt_f16_kernel<<<(DM * DM) / 256, 256, 0, stream>>>(Wv, Wvh, DM * DM);
  cvt_f16_kernel<<<(DM * DM) / 256, 256, 0, stream>>>(Wo, Woh, DM * DM);
  cvt_f16_kernel<<<(DG * DM) / 256, 256, 0, stream>>>(Wg, Wgh, DG * DM);

  // projections: 16x64 strip per wave, f16 outputs
  {
    int waves = (NT / 16) * (DM / 64);  // 1024
    gemm4_kernel<_Float16><<<waves / 8, 256, 0, stream>>>(xh, Wqh, qh, NT, DM, DM);
    gemm4_kernel<_Float16><<<waves / 8, 256, 0, stream>>>(xh, Wkh, kh, NT, DM, DM);
    gemm4_kernel<_Float16><<<waves / 8, 256, 0, stream>>>(xh, Wvh, vh, NT, DM, DM);
    int gwaves = (NT / 16) * (DG / 16); // 128, N=16 -> single-tile kernel
    gemm1_kernel<<<gwaves / 8, 256, 0, stream>>>(xh, Wgh, gl, NT, DG, DM);
  }

  kcompress_kernel<<<NG, DM, 0, stream>>>(kh, kc);

  select_top2_kernel<<<(NH * NT) / 8, 256, 0, stream>>>(qh, kc, ix);
  inter_kernel<<<(NH * NT) / 8, 256, 0, stream>>>(qh, kh, vh, ix, oi);

  intra_kernel<<<NG * NH, 256, 0, stream>>>(qh, kh, vh, oa);

  combine_kernel<<<(NT * DM) / 256, 256, 0, stream>>>(oi, oa, gl, oh);

  // out = o @ Wo^T (f32 to d_out)
  {
    int waves = (NT / 16) * (DM / 64);
    gemm4_kernel<float><<<waves / 8, 256, 0, stream>>>(oh, Woh, out, NT, DM, DM);
  }
  (void)in_sizes; (void)n_in; (void)out_size; (void)ws_size;
}
